// Transformer_40286793236910
// MI455X (gfx1250) — matrix operand order repaired and checked by reading, not compiled, not run on hardware
//
#include <hip/hip_runtime.h>
#include <math.h>

#define NN 16384
#define EE 262144
#define INV_SQRT3f 0.57735026918962576f
#define GSCALE 0.125f       // 1/sqrt(64)
#define LSCALE 0.03125f     // 1/sqrt(1024)
#define OSCALE 0.1767766952966369f // 1/sqrt(32)

typedef _Float16 f16;
typedef __attribute__((ext_vector_type(16))) _Float16 v16h;
typedef __attribute__((ext_vector_type(8)))  _Float16 v8h;
typedef __attribute__((ext_vector_type(8)))  float    v8f;

// ---------------- helpers ----------------

__device__ __forceinline__ float fast_tanh(float x) {
#if __has_builtin(__builtin_amdgcn_tanhf)
    return __builtin_amdgcn_tanhf(x);   // native V_TANH_F32 on gfx1250
#else
    return tanhf(x);
#endif
}

__device__ __forceinline__ float gelu_tanh(float x) {
    float x3 = x * x * x;
    return 0.5f * x * (1.0f + fast_tanh(0.7978845608028654f * (x + 0.044715f * x3)));
}

__device__ __forceinline__ unsigned fenc(float x) {
    unsigned u = __float_as_uint(x);
    return (u & 0x80000000u) ? ~u : (u | 0x80000000u);
}
__device__ __forceinline__ float fdec(unsigned k) {
    return (k & 0x80000000u) ? __uint_as_float(k ^ 0x80000000u)
                             : __uint_as_float(~k);
}

__device__ __forceinline__ v8f zero8() {
    v8f z;
#pragma unroll
    for (int i = 0; i < 8; ++i) z[i] = 0.0f;
    return z;
}

__device__ __forceinline__ v8f wmma_f16(v16h a, v16h b, v8f c) {
    return __builtin_amdgcn_wmma_f32_16x16x32_f16(false, a, false, b,
                                                  (short)0, c, false, false);
}

// A fragment (16xK tile, 16-bit): lane m=lane&15, hi=lane>>4.
// element e<8  -> K = kb + hi*8 + e
// element e>=8 -> K = kb + 16 + hi*8 + (e-8)
__device__ __forceinline__ v16h load_a_frag(const f16* buf, int rowStride,
                                            int kb, int lane) {
    int m = lane & 15, hi = lane >> 4;
    const v8h r1 = *(const v8h*)(buf + m * rowStride + kb + hi * 8);
    const v8h r2 = *(const v8h*)(buf + m * rowStride + kb + 16 + hi * 8);
    v16h a;
#pragma unroll
    for (int i = 0; i < 8; ++i) { a[i] = r1[i]; a[i + 8] = r2[i]; }
    return a;
}

// B fragment from transposed weights Wt[n][k] (k contiguous):
// element e<8 -> B[k = kt*32 + hi*8 + e][n], e>=8 -> k = kt*32 + 16 + hi*8 + (e-8); n = (lane&15)+nt*16
__device__ __forceinline__ v16h load_b_frag(const f16* wt, int rowStride,
                                            int kt, int nt, int lane) {
    int n = (lane & 15) + nt * 16, hi = lane >> 4;
    const f16* base = wt + n * rowStride + kt * 32 + hi * 8;
    const v8h r1 = *(const v8h*)(base);
    const v8h r2 = *(const v8h*)(base + 16);
    v16h b;
#pragma unroll
    for (int i = 0; i < 8; ++i) { b[i] = r1[i]; b[i + 8] = r2[i]; }
    return b;
}

// C layout: lane col n = (lane&15)+nt*16, element r -> row m = r + 8*(lane>>4)
__device__ __forceinline__ void store_c_f16(f16* buf, int rowStride, int nt,
                                            int lane, v8f acc) {
    int n = (lane & 15) + nt * 16, hi = lane >> 4;
#pragma unroll
    for (int r = 0; r < 8; ++r)
        buf[(r + hi * 8) * rowStride + n] = (f16)acc[r];
}

// 3-layer-MLP tail: hA holds gelu(y_s*W0) (16x64 f16, row stride 72).
// Computes g = gelu(hA@W1/8)@W2/8 into gOut (f16, row stride 72).
__device__ __forceinline__ void run_mlp(f16* hA, f16* gOut,
                                        const f16* W1t, const f16* W2t,
                                        int lane) {
    const int n15 = lane & 15, hi = lane >> 4;
    {   // layer 1 (gelu)
        v16h a0 = load_a_frag(hA, 72, 0, lane);
        v16h a1 = load_a_frag(hA, 72, 32, lane);
#pragma unroll
        for (int nt = 0; nt < 4; ++nt) {
            v8f acc = zero8();
            acc = wmma_f16(a0, load_b_frag(W1t, 72, 0, nt, lane), acc);
            acc = wmma_f16(a1, load_b_frag(W1t, 72, 1, nt, lane), acc);
            int col = nt * 16 + n15;
#pragma unroll
            for (int r = 0; r < 8; ++r)
                hA[(r + hi * 8) * 72 + col] = (f16)gelu_tanh(acc[r] * GSCALE);
        }
    }
    __syncthreads();
    {   // layer 2 (no gelu)
        v16h a0 = load_a_frag(hA, 72, 0, lane);
        v16h a1 = load_a_frag(hA, 72, 32, lane);
#pragma unroll
        for (int nt = 0; nt < 4; ++nt) {
            v8f acc = zero8();
            acc = wmma_f16(a0, load_b_frag(W2t, 72, 0, nt, lane), acc);
            acc = wmma_f16(a1, load_b_frag(W2t, 72, 1, nt, lane), acc);
            int col = nt * 16 + n15;
#pragma unroll
            for (int r = 0; r < 8; ++r)
                gOut[(r + hi * 8) * 72 + col] = (f16)(acc[r] * GSCALE);
        }
    }
}

// ---------------- kernels ----------------

__global__ void k_init(unsigned* __restrict__ lmaxEnc, float* __restrict__ zreg,
                       int nLmax, long long nZero) {
    long long idx = (long long)blockIdx.x * 256 + threadIdx.x;
    if (idx < nLmax) lmaxEnc[idx] = 0x80000000u; // enc(+0.0f) (ref seeds max with 0)
    if (idx < nZero) zreg[idx] = 0.0f;
}

// Pass 1: key MLP + logit path (WMMA), store logits, atomic-max per dst.
__global__ __launch_bounds__(64) void k_edge_logits(
    const int* __restrict__ edge_src, const int* __restrict__ edge_dst,
    const float* __restrict__ edge_attr, const float* __restrict__ node_feat,
    const float* __restrict__ W0, const float* __restrict__ W1,
    const float* __restrict__ W2, const float* __restrict__ WL,
    float* __restrict__ logit_out, unsigned* __restrict__ lmaxEnc) {
    __shared__ alignas(16) f16 W1t[64][72];
    __shared__ alignas(16) f16 W2t[64][72];
    __shared__ alignas(16) f16 T1t[64][40];
    __shared__ alignas(16) f16 T2t[64][40];
    __shared__ alignas(16) f16 hA[2][16][72];
    __shared__ alignas(16) f16 Cb[2][16][72];
    __shared__ alignas(16) f16 sF[2][16][72];
    __shared__ alignas(16) f16 ksB[2][16][40];
    __shared__ alignas(16) f16 kvB[2][3][16][40];
    __shared__ float yv4[2][16][4];
    __shared__ int dstI[2][16], srcI[2][16];

    const int tid = threadIdx.x, lane = tid & 31, w = tid >> 5;
    const int e0 = blockIdx.x * 32 + w * 16;

    for (int idx = tid; idx < 4096; idx += 64) {
        int n = idx & 63, k = idx >> 6;
        W1t[n][k] = (f16)W1[k * 64 + n];
        W2t[n][k] = (f16)W2[k * 64 + n];
    }
    for (int idx = tid; idx < 2048; idx += 64) {
        int col = idx >> 5, j = idx & 31;
        int i = col >> 2, h = col & 3;
        T1t[col][j] = (f16)WL[(i * 32 + j) * 4 + h];
        T2t[col][j] = (f16)WL[(512 + i * 32 + j) * 4 + h];
    }
    if (lane < 16) {
        int e = e0 + lane;
        dstI[w][lane] = edge_dst[e];
        srcI[w][lane] = edge_src[e];
        yv4[w][lane][0] = edge_attr[e * 4 + 1];
        yv4[w][lane][1] = edge_attr[e * 4 + 2];
        yv4[w][lane][2] = edge_attr[e * 4 + 3];
        yv4[w][lane][3] = edge_attr[e * 4 + 0]; // y_s
    }
    __syncthreads();

    // h0 = gelu(y_s * W0) and stage src features
    for (int idx = lane; idx < 1024; idx += 32) {
        int m = idx >> 6, j = idx & 63;
        hA[w][m][j] = (f16)gelu_tanh(yv4[w][m][3] * W0[j]);
        sF[w][m][j] = (f16)node_feat[(size_t)srcI[w][m] * 64 + j];
    }
    __syncthreads();

    run_mlp(&hA[w][0][0], &Cb[w][0][0], &W1t[0][0], &W2t[0][0], lane);
    __syncthreads();

    // ks tile (16x32): s = [xs, INV_SQRT3 * xv.y_v] * g[:32]
    for (int idx = lane; idx < 512; idx += 32) {
        int m = idx >> 5, j = idx & 31;
        float g = (float)Cb[w][m][j];
        float s;
        if (j < 16) {
            s = (float)sF[w][m][j];
        } else {
            int i = j - 16;
            s = INV_SQRT3f * ((float)sF[w][m][16 + i * 3 + 0] * yv4[w][m][0] +
                              (float)sF[w][m][16 + i * 3 + 1] * yv4[w][m][1] +
                              (float)sF[w][m][16 + i * 3 + 2] * yv4[w][m][2]);
        }
        ksB[w][m][j] = (f16)(s * g);
    }
    // kv tiles (3 x 16x32): v = [xv, xs (x) y_v] * g[32:]
    for (int idx = lane; idx < 1536; idx += 32) {
        int m = idx / 96, u = idx % 96, r = u / 3, c = u % 3;
        float g = (float)Cb[w][m][32 + r];
        float vc = (r < 16) ? (float)sF[w][m][16 + r * 3 + c]
                            : (float)sF[w][m][r - 16] * yv4[w][m][c];
        kvB[w][c][m][r] = (f16)(vc * g);
    }
    __syncthreads();

    const int mm = lane & 15, hh = lane >> 4;       // lane owns (edge mm, heads hh, hh+2)
    const float* drow = node_feat + (size_t)dstI[w][mm] * 64;
    float lac0 = 0.0f, lac1 = 0.0f;

    {   // part 1: a = ks @ T1 ; logit += ds . a
        v16h a = load_a_frag(&ksB[w][0][0], 40, 0, lane);
#pragma unroll
        for (int nt = 0; nt < 4; ++nt) {
            v8f acc = zero8();
            acc = wmma_f16(a, load_b_frag(&T1t[0][0], 40, 0, nt, lane), acc);
            store_c_f16(&Cb[w][0][0], 72, nt, lane, acc);
        }
    }
    __syncthreads();
    {
        float s0 = 0.0f, s1 = 0.0f;
#pragma unroll
        for (int i = 0; i < 16; ++i) {
            float d = drow[i];
            s0 += d * (float)Cb[w][mm][i * 4 + hh];
            s1 += d * (float)Cb[w][mm][i * 4 + hh + 2];
        }
        lac0 += s0; lac1 += s1;
    }
    __syncthreads();

    for (int c = 0; c < 3; ++c) {  // part 2: b_c = kv_c @ T2 ; logit += dv_c . b_c / sqrt3
        v16h a = load_a_frag(&kvB[w][c][0][0], 40, 0, lane);
#pragma unroll
        for (int nt = 0; nt < 4; ++nt) {
            v8f acc = zero8();
            acc = wmma_f16(a, load_b_frag(&T2t[0][0], 40, 0, nt, lane), acc);
            store_c_f16(&Cb[w][0][0], 72, nt, lane, acc);
        }
        __syncthreads();
        float s0 = 0.0f, s1 = 0.0f;
#pragma unroll
        for (int i = 0; i < 16; ++i) {
            float d = drow[16 + i * 3 + c];
            s0 += d * (float)Cb[w][mm][i * 4 + hh];
            s1 += d * (float)Cb[w][mm][i * 4 + hh + 2];
        }
        lac0 += INV_SQRT3f * s0; lac1 += INV_SQRT3f * s1;
        __syncthreads();
    }

    int e = e0 + mm;
    float l0 = lac0 * LSCALE, l1 = lac1 * LSCALE;
    logit_out[e * 4 + hh] = l0;
    logit_out[e * 4 + hh + 2] = l1;
    atomicMax(&lmaxEnc[dstI[w][mm] * 4 + hh], fenc(l0));
    atomicMax(&lmaxEnc[dstI[w][mm] * 4 + hh + 2], fenc(l1));
}

// Pass 2: ex = cutoff * exp(logit - lmax[dst]); z[dst] += ex  (in-place on buffer)
__global__ void k_softmax_z(const int* __restrict__ edge_dst,
                            const float* __restrict__ cutoff,
                            float* __restrict__ exbuf,
                            const unsigned* __restrict__ lmaxEnc,
                            float* __restrict__ zbuf) {
    int idx = blockIdx.x * 256 + threadIdx.x;
    int e = idx >> 2, h = idx & 3;
    int d = edge_dst[e];
    float lm = fdec(lmaxEnc[d * 4 + h]);
    float ex = cutoff[e] * __expf(exbuf[idx] - lm);  // arg <= 0, native v_exp path
    exbuf[idx] = ex;
    atomicAdd(&zbuf[d * 4 + h], ex);
}

// Pass 3: val MLP (WMMA) + weighted scatter-add into ns / nv.
__global__ __launch_bounds__(64) void k_edge_values(
    const int* __restrict__ edge_src, const int* __restrict__ edge_dst,
    const float* __restrict__ edge_attr, const float* __restrict__ node_feat,
    const float* __restrict__ W0, const float* __restrict__ W1,
    const float* __restrict__ W2, const float* __restrict__ exbuf,
    const float* __restrict__ zbuf, float* __restrict__ nsb,
    float* __restrict__ nvb) {
    __shared__ alignas(16) f16 W1t[64][72];
    __shared__ alignas(16) f16 W2t[64][72];
    __shared__ alignas(16) f16 hA[2][16][72];
    __shared__ alignas(16) f16 Cb[2][16][72];
    __shared__ alignas(16) f16 sF[2][16][72];
    __shared__ float yv4[2][16][4];
    __shared__ float wgt[2][16][4];
    __shared__ int dstI[2][16], srcI[2][16];

    const int tid = threadIdx.x, lane = tid & 31, w = tid >> 5;
    const int e0 = blockIdx.x * 32 + w * 16;

    for (int idx = tid; idx < 4096; idx += 64) {
        int n = idx & 63, k = idx >> 6;
        W1t[n][k] = (f16)W1[k * 64 + n];
        W2t[n][k] = (f16)W2[k * 64 + n];
    }
    if (lane < 16) {
        int e = e0 + lane;
        dstI[w][lane] = edge_dst[e];
        srcI[w][lane] = edge_src[e];
        yv4[w][lane][0] = edge_attr[e * 4 + 1];
        yv4[w][lane][1] = edge_attr[e * 4 + 2];
        yv4[w][lane][2] = edge_attr[e * 4 + 3];
        yv4[w][lane][3] = edge_attr[e * 4 + 0];
    }
    __syncthreads();

    for (int idx = lane; idx < 1024; idx += 32) {
        int m = idx >> 6, j = idx & 63;
        hA[w][m][j] = (f16)gelu_tanh(yv4[w][m][3] * W0[j]);
        sF[w][m][j] = (f16)node_feat[(size_t)srcI[w][m] * 64 + j];
    }
    __syncthreads();

    run_mlp(&hA[w][0][0], &Cb[w][0][0], &W1t[0][0], &W2t[0][0], lane);
    __syncthreads();

    // attention weights: w = sqrt(relu(ex / z')), z'=1 where z==0
    for (int idx = lane; idx < 64; idx += 32) {
        int m = idx >> 2, h = idx & 3;
        float zz = zbuf[dstI[w][m] * 4 + h];
        zz = (zz == 0.0f) ? 1.0f : zz;
        float al = exbuf[(e0 + m) * 4 + h] / zz;
        wgt[w][m][h] = sqrtf(fmaxf(al, 0.0f));
    }
    __syncthreads();

    // vs: (E,32) grouped as (4 heads x 8)
    for (int idx = lane; idx < 512; idx += 32) {
        int m = idx >> 5, q = idx & 31;
        float g = (float)Cb[w][m][q];
        float s;
        if (q < 16) {
            s = (float)sF[w][m][q];
        } else {
            int i = q - 16;
            s = INV_SQRT3f * ((float)sF[w][m][16 + i * 3 + 0] * yv4[w][m][0] +
                              (float)sF[w][m][16 + i * 3 + 1] * yv4[w][m][1] +
                              (float)sF[w][m][16 + i * 3 + 2] * yv4[w][m][2]);
        }
        float val = s * g * wgt[w][m][q >> 3];
        atomicAdd(&nsb[(size_t)dstI[w][m] * 32 + q], val);
    }
    // vv: (E,32,3) grouped as (4 heads x 8, 3)
    for (int idx = lane; idx < 1536; idx += 32) {
        int m = idx / 96, u = idx % 96, r = u / 3, c = u % 3;
        float g = (float)Cb[w][m][32 + r];
        float vc = (r < 16) ? (float)sF[w][m][16 + r * 3 + c]
                            : (float)sF[w][m][r - 16] * yv4[w][m][c];
        float val = vc * g * wgt[w][m][r >> 3];
        atomicAdd(&nvb[((size_t)dstI[w][m] * 32 + r) * 3 + c], val);
    }
}

// Pass 4: per-node projections (WMMA), interleaved [out_s | out_v] output.
__global__ __launch_bounds__(64) void k_node_out(
    const float* __restrict__ nsb, const float* __restrict__ nvb,
    const float* __restrict__ wos, const float* __restrict__ wov,
    float* __restrict__ out) {
    __shared__ alignas(16) f16 WsT[16][40];
    __shared__ alignas(16) f16 WvT[16][40];
    __shared__ alignas(16) f16 aS[2][16][40];

    const int tid = threadIdx.x, lane = tid & 31, w = tid >> 5;
    const int n0 = blockIdx.x * 32 + w * 16;

    for (int idx = tid; idx < 512; idx += 64) {
        int o = idx >> 5, r = idx & 31;
        WsT[o][r] = (f16)wos[r * 16 + o];
        WvT[o][r] = (f16)wov[r * 16 + o];
    }
    for (int idx = lane; idx < 512; idx += 32) {
        int m = idx >> 5, r = idx & 31;
        aS[w][m][r] = (f16)nsb[(size_t)(n0 + m) * 32 + r];
    }
    __syncthreads();

    const int n15 = lane & 15, hi = lane >> 4;
    {   // out_s = ns @ w_out_s / sqrt(32)
        v16h a = load_a_frag(&aS[w][0][0], 40, 0, lane);
        v8f acc = zero8();
        acc = wmma_f16(a, load_b_frag(&WsT[0][0], 40, 0, 0, lane), acc);
#pragma unroll
        for (int r = 0; r < 8; ++r)
            out[(size_t)(n0 + r + hi * 8) * 64 + n15] = acc[r] * OSCALE;
    }
    for (int c = 0; c < 3; ++c) {  // out_v[:, o, c] = nv[:, :, c] @ w_out_v / sqrt(32)
        __syncthreads();
        for (int idx = lane; idx < 512; idx += 32) {
            int m = idx >> 5, r = idx & 31;
            aS[w][m][r] = (f16)nvb[((size_t)(n0 + m) * 32 + r) * 3 + c];
        }
        __syncthreads();
        v16h a = load_a_frag(&aS[w][0][0], 40, 0, lane);
        v8f acc = zero8();
        acc = wmma_f16(a, load_b_frag(&WvT[0][0], 40, 0, 0, lane), acc);
#pragma unroll
        for (int r = 0; r < 8; ++r)
            out[(size_t)(n0 + r + hi * 8) * 64 + 16 + n15 * 3 + c] =
                acc[r] * OSCALE;
    }
}

// ---------------- launcher ----------------

extern "C" void kernel_launch(void* const* d_in, const int* in_sizes, int n_in,
                              void* d_out, int out_size, void* d_ws,
                              size_t ws_size, hipStream_t stream) {
    (void)in_sizes; (void)n_in; (void)out_size; (void)ws_size;
    const int*   edge_src  = (const int*)d_in[0];
    const int*   edge_dst  = (const int*)d_in[1];
    const float* cutoff    = (const float*)d_in[2];
    const float* edge_attr = (const float*)d_in[3];
    const float* node_feat = (const float*)d_in[4];
    const float* wk0 = (const float*)d_in[5];
    const float* wk1 = (const float*)d_in[6];
    const float* wk2 = (const float*)d_in[7];
    const float* wv0 = (const float*)d_in[8];
    const float* wv1 = (const float*)d_in[9];
    const float* wv2 = (const float*)d_in[10];
    const float* wl  = (const float*)d_in[11];
    const float* wos = (const float*)d_in[12];
    const float* wov = (const float*)d_in[13];
    float* out = (float*)d_out;

    // workspace layout (f32 words): ex/logit[E*4] | lmaxEnc[N*4] | z[N*4] | ns[N*32] | nv[N*96]
    float*    ws      = (float*)d_ws;
    float*    exbuf   = ws;
    unsigned* lmaxEnc = (unsigned*)(ws + (size_t)EE * 4);
    float*    zbuf    = ws + (size_t)EE * 4 + (size_t)NN * 4;
    float*    nsb     = zbuf + (size_t)NN * 4;
    float*    nvb     = nsb + (size_t)NN * 32;

    long long nZero = (long long)NN * (4 + 32 + 96); // z + ns + nv contiguous
    int initBlocks = (int)((nZero + 255) / 256);
    k_init<<<initBlocks, 256, 0, stream>>>(lmaxEnc, zbuf, NN * 4, nZero);

    k_edge_logits<<<EE / 32, 64, 0, stream>>>(edge_src, edge_dst, edge_attr,
                                              node_feat, wk0, wk1, wk2, wl,
                                              exbuf, lmaxEnc);

    k_softmax_z<<<(EE * 4) / 256, 256, 0, stream>>>(edge_dst, cutoff, exbuf,
                                                    lmaxEnc, zbuf);

    k_edge_values<<<EE / 32, 64, 0, stream>>>(edge_src, edge_dst, edge_attr,
                                              node_feat, wv0, wv1, wv2, exbuf,
                                              zbuf, nsb, nvb);

    k_node_out<<<NN / 32, 64, 0, stream>>>(nsb, nvb, wos, wov, out);
}
